// Indexer_61890478735734
// MI455X (gfx1250) — matrix operand order repaired and checked by reading, not compiled, not run on hardware
//
#include <hip/hip_runtime.h>
#include <hip/hip_bf16.h>

typedef __attribute__((ext_vector_type(16))) __bf16 v16bf;
typedef __attribute__((ext_vector_type(8)))  float  v8f;

constexpr int kB      = 32;
constexpr int kQLEN   = 4;
constexpr int kKV     = 8192;
constexpr int kBLK    = 128;
constexpr int kNHEAD  = 64;
constexpr int kHD     = 128;
constexpr int kROPE   = 64;
constexpr int kTOPK   = 2048;
constexpr int kHIDDEN = 7168;
constexpr int kQLORA  = 1536;
constexpr int kT      = kB * kQLEN;       // 128
constexpr int kQN     = kNHEAD * kHD;     // 8192

static __device__ inline __bf16 f2bf(float x) { return (__bf16)x; }

static __device__ inline unsigned pack2bf(float a, float b) {
  __bf16 ba = (__bf16)a, bb = (__bf16)b;
  unsigned short lo, hi;
  __builtin_memcpy(&lo, &ba, 2);
  __builtin_memcpy(&hi, &bb, 2);
  return (unsigned)lo | ((unsigned)hi << 16);
}

// A fragment: 16x32 bf16, row-major source [m][k]; lanes 0-15 carry K 0..15,
// lanes 16-31 carry K 16..31 (pairs per VGPR).
static __device__ inline v16bf load_a_frag(const __bf16* row, int lane) {
  int ko = (lane & 16) ? 16 : 0;
  v16bf f;
  __builtin_memcpy(&f, row + ko, 16);
  __builtin_memcpy(reinterpret_cast<char*>(&f) + 16, row + ko + 8, 16);
  return f;
}

// B fragment: 32x16 bf16 with B[k][n] = src[n][k]. Lanes 0-15: K=0..15,
// lanes 16-31: K=16..31 -> one contiguous 32B load per lane.
static __device__ inline v16bf load_b_frag(const __bf16* row, int lane) {
  int ko = (lane & 16) ? 16 : 0;
  v16bf f;
  __builtin_memcpy(&f, row + ko, 32);
  return f;
}

static __device__ inline v8f wmma_bf16(v16bf a, v16bf b, v8f c) {
  return __builtin_amdgcn_wmma_f32_16x16x32_bf16(false, a, false, b, (short)0, c,
                                                 false, false);
}

// ---------------------------------------------------------------------------
// K0: k = LN(hidden @ wk) with RoPE on first 64 dims. One block per token row.
// ---------------------------------------------------------------------------
__global__ void k_kernel(const float* __restrict__ hidden,
                         const float* __restrict__ wk,
                         const float* __restrict__ kns,
                         const float* __restrict__ knb,
                         const float* __restrict__ cosb,
                         const float* __restrict__ sinb,
                         float* __restrict__ kout,
                         __bf16* __restrict__ knew) {
  const int t = blockIdx.x;
  const int d = threadIdx.x;   // 128 threads
  __shared__ float hbuf[512];
  __shared__ float kv[kHD];
  __shared__ float kvn[kHD];
  __shared__ float smu, svar;

  float acc = 0.f;
  for (int h0 = 0; h0 < kHIDDEN; h0 += 512) {
#pragma unroll
    for (int i = 0; i < 4; ++i)
      hbuf[d + 128 * i] = hidden[(size_t)t * kHIDDEN + h0 + d + 128 * i];
    __syncthreads();
    for (int h = 0; h < 512; ++h)
      acc = fmaf(hbuf[h], wk[(size_t)(h0 + h) * kHD + d], acc);
    __syncthreads();
  }
  kv[d] = acc;
  __syncthreads();
  if (d == 0) {
    float mu = 0.f;
    for (int i = 0; i < kHD; ++i) mu += kv[i];
    mu *= (1.0f / kHD);
    float var = 0.f;
    for (int i = 0; i < kHD; ++i) { float c = kv[i] - mu; var += c * c; }
    smu = mu; svar = var * (1.0f / kHD);
  }
  __syncthreads();
  float kn = (acc - smu) * rsqrtf(svar + 1e-6f) * kns[d] + knb[d];
  kvn[d] = kn;
  __syncthreads();
  float out = kn;
  if (d < kROPE) {
    float c = cosb[t * kROPE + d], s = sinb[t * kROPE + d];
    float p = kvn[d ^ 32];
    out = kn * c + ((d < 32) ? -p : p) * s;
  }
  kout[t * kHD + d] = out;
  knew[t * kHD + d] = f2bf(out);
}

// ---------------------------------------------------------------------------
// K0b: weights = hidden @ weights_proj_w + bias. One block per token row.
// ---------------------------------------------------------------------------
__global__ void w_kernel(const float* __restrict__ hidden,
                         const float* __restrict__ wpw,
                         const float* __restrict__ wpb,
                         float* __restrict__ wws) {
  const int t = blockIdx.x;
  const int hh = threadIdx.x;  // 64 threads
  __shared__ float hbuf[256];
  float acc = 0.f;
  for (int h0 = 0; h0 < kHIDDEN; h0 += 256) {
#pragma unroll
    for (int i = 0; i < 4; ++i)
      hbuf[hh + 64 * i] = hidden[(size_t)t * kHIDDEN + h0 + hh + 64 * i];
    __syncthreads();
    for (int h = 0; h < 256; ++h)
      acc = fmaf(hbuf[h], wpw[(size_t)(h0 + h) * kNHEAD + hh], acc);
    __syncthreads();
  }
  wws[t * kNHEAD + hh] = acc + wpb[hh];
}

// ---------------------------------------------------------------------------
// Keys build: gather via block_table, override last Q_LEN rows with new k.
// 4 rows per block; float4 loads, packed 8-byte bf16 stores.
// ---------------------------------------------------------------------------
__global__ void keys_kernel(const float* __restrict__ cache,
                            const int* __restrict__ btab,
                            const __bf16* __restrict__ knew,
                            __bf16* __restrict__ keysb) {
  const int tid = threadIdx.x;              // 128
  const int row = blockIdx.x * 4 + (tid >> 5);
  const int d = (tid & 31) * 4;
  const int b = row >> 13;
  const int kpos = row & (kKV - 1);
  uint2 packed;
  if (kpos >= kKV - kQLEN) {
    const __bf16* src =
        knew + (b * kQLEN + (kpos - (kKV - kQLEN))) * kHD + d;
    __builtin_memcpy(&packed, src, 8);
  } else {
    int blk = btab[b * (kKV / kBLK) + (kpos >> 7)];
    const float4 v = *(const float4*)(cache +
        (((size_t)blk * kBLK + (kpos & 127)) * kHD + d));
    packed.x = pack2bf(v.x, v.y);
    packed.y = pack2bf(v.z, v.w);
  }
  *(uint2*)(keysb + (size_t)row * kHD + d) = packed;
}

// ---------------------------------------------------------------------------
// K1: q_raw = qr @ wq_b, bf16 WMMA. M=128, N=8192, K=1536.
// Grid: 64 blocks (N tiles of 128), 256 threads = 8 waves; wave w owns n-subtile w.
// Interleaved A-frag load -> WMMA (no frag array: keeps VGPR pressure below
// the spill threshold; scheduler overlaps ds_load pairs with the XDL pipe).
// ---------------------------------------------------------------------------
__global__ void qgemm_kernel(const float* __restrict__ qr,
                             const float* __restrict__ wq_b,
                             float* __restrict__ q_raw) {
  const int n0 = blockIdx.x * 128;
  const int tid = threadIdx.x;
  const int lane = tid & 31;
  const int wave = tid >> 5;
  __shared__ unsigned Asu[128 * 32];  // [m][kpair], K_TILE=64
  __shared__ __bf16 Bs[128 * 64];     // [n][k]

  v8f acc[8];
#pragma unroll
  for (int i = 0; i < 8; ++i) acc[i] = v8f{0, 0, 0, 0, 0, 0, 0, 0};

  for (int k0 = 0; k0 < kQLORA; k0 += 64) {
#pragma unroll
    for (int i = 0; i < 16; ++i) {
      int p = tid + 256 * i;
      int m = p >> 5, kp = p & 31;
      const float2 v = *(const float2*)(qr + (size_t)m * kQLORA + k0 + 2 * kp);
      Asu[m * 32 + kp] = pack2bf(v.x, v.y);
    }
#pragma unroll
    for (int i = 0; i < 32; ++i) {
      int idx = tid + 256 * i;
      int n = idx & 127, k = idx >> 7;
      Bs[n * 64 + k] = f2bf(wq_b[(size_t)(k0 + k) * kQN + n0 + n]);
    }
    __syncthreads();
    const __bf16* As = (const __bf16*)Asu;
#pragma unroll
    for (int kk = 0; kk < 2; ++kk) {
      v16bf bf = load_b_frag(&Bs[(16 * wave + (lane & 15)) * 64 + 32 * kk], lane);
#pragma unroll
      for (int mt = 0; mt < 8; ++mt) {
        v16bf af = load_a_frag(&As[(16 * mt + (lane & 15)) * 64 + 32 * kk], lane);
        acc[mt] = wmma_bf16(af, bf, acc[mt]);
      }
    }
    __syncthreads();
  }
  const int col = n0 + 16 * wave + (lane & 15);
  const int mofs = (lane & 16) ? 8 : 0;
#pragma unroll
  for (int mt = 0; mt < 8; ++mt)
#pragma unroll
    for (int r = 0; r < 8; ++r)
      q_raw[(size_t)(16 * mt + r + mofs) * kQN + col] = acc[mt][r];
}

// ---------------------------------------------------------------------------
// K1b: RoPE q's first 64 dims per head; repack bf16 as qarr[b][m=h*4+q][d].
// 64 threads, each owns an even/odd dim pair -> one packed 32-bit store.
// ---------------------------------------------------------------------------
__global__ void qrope_kernel(const float* __restrict__ q_raw,
                             const float* __restrict__ cosb,
                             const float* __restrict__ sinb,
                             __bf16* __restrict__ qarr) {
  const int t = blockIdx.x;     // token
  const int j = threadIdx.x;    // 64
  const int d0 = 2 * j, d1 = 2 * j + 1;
  const int b = t >> 2, qi = t & 3;
  float c0 = 0.f, s0 = 0.f, c1 = 0.f, s1 = 0.f;
  if (d0 < kROPE) {
    c0 = cosb[t * kROPE + d0]; s0 = sinb[t * kROPE + d0];
    c1 = cosb[t * kROPE + d1]; s1 = sinb[t * kROPE + d1];
  }
  for (int h = 0; h < kNHEAD; ++h) {
    const float* base = q_raw + (size_t)t * kQN + h * kHD;
    float x0 = base[d0], x1 = base[d1];
    float o0 = x0, o1 = x1;
    if (d0 < kROPE) {
      float p0 = base[d0 ^ 32], p1 = base[d1 ^ 32];
      o0 = x0 * c0 + ((d0 < 32) ? -p0 : p0) * s0;
      o1 = x1 * c1 + ((d1 < 32) ? -p1 : p1) * s1;
    }
    int m = h * kQLEN + qi;
    *(unsigned*)(qarr + ((size_t)b * 256 + m) * kHD + d0) = pack2bf(o0, o1);
  }
}

// ---------------------------------------------------------------------------
// K2: score[b][q][k] = sum_h w[b,q,h]*relu(q[b,q,h,:]·keys[b,k,:]) + causal.
// Wave owns 2 n-tiles, sweeps 16 m-tiles; two independent WMMA accumulators
// per m-tile; head reduction in registers (q = r&3 invariant), halves merged
// with shfl_xor(16).
// ---------------------------------------------------------------------------
__global__ void score_kernel(const __bf16* __restrict__ qarr,
                             const __bf16* __restrict__ keysb,
                             const float* __restrict__ wws,
                             float* __restrict__ score) {
  const int chunk = blockIdx.x;   // 0..31 -> 256 keys each
  const int b = blockIdx.y;
  const int tid = threadIdx.x;
  const int lane = tid & 31;
  const int wave = tid >> 5;
  __shared__ float wlds[256];
  if (tid < 256) {
    int m = tid;
    wlds[m] = wws[(b * kQLEN + (m & 3)) * kNHEAD + (m >> 2)];
  }

  const __bf16* qA = qarr + (size_t)b * 256 * kHD;
  const __bf16* kB = keysb + (size_t)b * kKV * kHD;

  // prefetch this wave's second n-tile rows (gfx1250 global_prefetch path)
  {
    int nb2 = chunk * 256 + (wave * 2 + 1) * 16 + (lane & 15);
    __builtin_prefetch(&kB[(size_t)nb2 * kHD], 0, 1);
  }
  __syncthreads();

  for (int j = 0; j < 2; ++j) {
    const int nt = wave * 2 + j;
    const int nbase = chunk * 256 + nt * 16;
    const int nrow = nbase + (lane & 15);
    v16bf bf[4];
#pragma unroll
    for (int kk = 0; kk < 4; ++kk)
      bf[kk] = load_b_frag(&kB[(size_t)nrow * kHD + 32 * kk], lane);

    float vacc[8];
#pragma unroll
    for (int r = 0; r < 8; ++r) vacc[r] = 0.f;

    for (int mt = 0; mt < 16; ++mt) {
      const __bf16* arow = qA + (size_t)(16 * mt + (lane & 15)) * kHD;
      v8f d0 = v8f{0, 0, 0, 0, 0, 0, 0, 0};
      v8f d1 = v8f{0, 0, 0, 0, 0, 0, 0, 0};
      d0 = wmma_bf16(load_a_frag(arow + 0, lane),  bf[0], d0);
      d1 = wmma_bf16(load_a_frag(arow + 32, lane), bf[1], d1);
      d0 = wmma_bf16(load_a_frag(arow + 64, lane), bf[2], d0);
      d1 = wmma_bf16(load_a_frag(arow + 96, lane), bf[3], d1);
      const int mbase = 16 * mt + ((lane & 16) ? 8 : 0);
#pragma unroll
      for (int r = 0; r < 8; ++r) {
        float sv = d0[r] + d1[r];
        sv = sv > 0.f ? sv : 0.f;
        vacc[r] = fmaf(wlds[mbase + r], sv, vacc[r]);
      }
    }
#pragma unroll
    for (int q = 0; q < 4; ++q) {
      float s = vacc[q] + vacc[q + 4];
      s += __shfl_xor(s, 16, 32);
      if (lane < 16) {
        int kpos = nbase + lane;
        int qpos = kKV - kQLEN + q;
        float out = (kpos <= qpos) ? s : -__builtin_inff();
        score[((size_t)(b * kQLEN + q)) * kKV + kpos] = out;
      }
    }
  }
}

// ---------------------------------------------------------------------------
// K3: top-2048 of 8192 per row via 3-pass radix select (12+12+8 bits) on the
// monotonic float key. One block (256 threads) per row; scores staged in LDS.
// ---------------------------------------------------------------------------
static __device__ inline unsigned mono_key(float f) {
  unsigned u = __float_as_uint(f);
  return (u & 0x80000000u) ? ~u : (u | 0x80000000u);
}

__global__ void topk_kernel(const float* __restrict__ score,
                            float* __restrict__ outv,
                            int* __restrict__ outi) {
  const int t = blockIdx.x;
  const int tid = threadIdx.x;   // 256
  __shared__ float svals[kKV];         // 32 KB
  __shared__ unsigned hist[4096];      // 16 KB
  __shared__ unsigned csum[256];
  __shared__ int s_bin, s_above;
  __shared__ int s_cntAbove, s_cntTies;

  for (int i = tid; i < kKV; i += 256) svals[i] = score[(size_t)t * kKV + i];
  __syncthreads();

  int K_rem = kTOPK;
  unsigned prefix = 0;
  unsigned thrKey = 0;
  for (int p = 0; p < 3; ++p) {
    const int nb = (p == 2) ? 256 : 4096;
    const int shift = (p == 0) ? 20 : (p == 1) ? 8 : 0;
    for (int i = tid; i < nb; i += 256) hist[i] = 0;
    __syncthreads();
#pragma unroll 4
    for (int i = 0; i < 32; ++i) {
      int idx = tid + 256 * i;
      unsigned key = mono_key(svals[idx]);
      bool match = (p == 0) || (p == 1 && (key >> 20) == prefix) ||
                   (p == 2 && (key >> 8) == prefix);
      if (match) atomicAdd(&hist[(key >> shift) & (unsigned)(nb - 1)], 1u);
    }
    __syncthreads();
    const int per = nb >> 8;
    unsigned sum = 0;
    for (int i = 0; i < per; ++i) sum += hist[tid * per + i];
    csum[tid] = sum;
    __syncthreads();
    if (tid == 0) {
      unsigned run = 0;
      for (int c = 255; c >= 0; --c) {
        unsigned s0 = csum[c];
        csum[c] = run;
        run += s0;
      }
    }
    __syncthreads();
    unsigned above = csum[tid];
    for (int bi = per - 1; bi >= 0; --bi) {
      int bin = tid * per + bi;
      unsigned cnt = hist[bin];
      if (above < (unsigned)K_rem && above + cnt >= (unsigned)K_rem) {
        s_bin = bin;
        s_above = (int)above;
      }
      above += cnt;
    }
    __syncthreads();
    if (p == 0)       prefix = (unsigned)s_bin;
    else if (p == 1)  prefix = (prefix << 12) | (unsigned)s_bin;
    else              thrKey = (prefix << 8) | (unsigned)s_bin;
    K_rem -= s_above;
    __syncthreads();
  }

  if (tid == 0) { s_cntAbove = 0; s_cntTies = 0; }
  __syncthreads();
  const int tieBase = kTOPK - K_rem;
#pragma unroll 4
  for (int i = 0; i < 32; ++i) {
    int idx = tid + 256 * i;
    float v = svals[idx];
    unsigned key = mono_key(v);
    if (key > thrKey) {
      int pos = atomicAdd(&s_cntAbove, 1);
      outv[(size_t)t * kTOPK + pos] = v;
      outi[(size_t)t * kTOPK + pos] = idx;
    } else if (key == thrKey) {
      int tp = atomicAdd(&s_cntTies, 1);
      if (tp < K_rem) {
        int pos = tieBase + tp;
        outv[(size_t)t * kTOPK + pos] = v;
        outi[(size_t)t * kTOPK + pos] = idx;
      }
    }
  }
}

// ---------------------------------------------------------------------------
extern "C" void kernel_launch(void* const* d_in, const int* in_sizes, int n_in,
                              void* d_out, int out_size, void* d_ws, size_t ws_size,
                              hipStream_t stream) {
  const float* hidden = (const float*)d_in[0];
  const float* qr     = (const float*)d_in[1];
  const float* cosb   = (const float*)d_in[2];
  const float* sinb   = (const float*)d_in[3];
  const float* wq_b   = (const float*)d_in[4];
  const float* wk     = (const float*)d_in[5];
  const float* kns    = (const float*)d_in[6];
  const float* knb    = (const float*)d_in[7];
  const float* wpw    = (const float*)d_in[8];
  const float* wpb    = (const float*)d_in[9];
  const float* cache  = (const float*)d_in[10];
  const int*   btab   = (const int*)d_in[11];
  (void)in_sizes; (void)n_in; (void)out_size; (void)ws_size;

  char* ws = (char*)d_ws;
  float*  q_raw = (float*)(ws);                                // 4 MB
  __bf16* qarr  = (__bf16*)(ws + (4u << 20));                  // 2 MB
  __bf16* knew  = (__bf16*)(ws + (6u << 20));                  // 32 KB
  float*  wws   = (float*)(ws + (6u << 20) + (64u << 10));     // 32 KB
  float*  scorep = (float*)(ws + (7u << 20));                  // 4 MB
  __bf16* keysb = (__bf16*)(ws + (11u << 20));                 // 64 MB

  float* outv = (float*)d_out;
  int*   outi = (int*)d_out + (size_t)kT * kTOPK;
  float* kout = (float*)d_out + 2 * (size_t)kT * kTOPK;

  k_kernel<<<kT, 128, 0, stream>>>(hidden, wk, kns, knb, cosb, sinb, kout, knew);
  w_kernel<<<kT, 64, 0, stream>>>(hidden, wpw, wpb, wws);
  keys_kernel<<<kB * kKV / 4, 128, 0, stream>>>(cache, btab, knew, keysb);
  qgemm_kernel<<<kQN / 128, 256, 0, stream>>>(qr, wq_b, q_raw);
  qrope_kernel<<<kT, 64, 0, stream>>>(q_raw, cosb, sinb, qarr);
  score_kernel<<<dim3(kKV / 256, kB), 256, 0, stream>>>(qarr, keysb, wws, scorep);
  topk_kernel<<<kT, 256, 0, stream>>>(scorep, outv, outi);
}
